// TrivialGNN_13365938225232
// MI455X (gfx1250) — compile-verified, matrix operand order checked by reading, not run on hardware
//
#include <hip/hip_runtime.h>
#include <hip/hip_bf16.h>
#include <stdint.h>

// CDNA5 / gfx1250: wave32, WMMA bf16 16x16x32 for dense GEMMs,
// global_load_async_to_lds_b128 for LDS staging.
typedef __attribute__((ext_vector_type(16))) __bf16 v16bf;
typedef __attribute__((ext_vector_type(8)))  __bf16 v8bf;
typedef __attribute__((ext_vector_type(8)))  float  v8f;

__device__ __forceinline__ uint32_t f2bfbits(float f) {
  // round-to-nearest-even f32 -> bf16 bit pattern (low 16 bits of result)
  uint32_t u = __builtin_bit_cast(uint32_t, f);
  u += 0x7FFFu + ((u >> 16) & 1u);
  return u >> 16;
}

__device__ __forceinline__ __bf16 f2bf(float f) {
  return __builtin_bit_cast(__bf16, (uint16_t)f2bfbits(f));
}

// Async copy 16B/lane from global to LDS (ASYNCcnt-tracked, CDNA5).
__device__ __forceinline__ void async_copy_b128(const void* gbase, unsigned gOff,
                                                unsigned ldsOff) {
  asm volatile("global_load_async_to_lds_b128 %0, %1, %2"
               :: "v"(ldsOff), "v"(gOff), "s"(gbase)
               : "memory");
}

__device__ __forceinline__ void wait_async0() {
  asm volatile("s_wait_asynccnt 0x0" ::: "memory");
}

__global__ void k_fill1(float* p, int n) {
  int i = blockIdx.x * blockDim.x + threadIdx.x;
  if (i < n) p[i] = 1.0f;
}

__global__ void k_zero4(float4* p, long n4) {
  long i = (long)blockIdx.x * blockDim.x + threadIdx.x;
  if (i < n4) p[i] = make_float4(0.f, 0.f, 0.f, 0.f);
}

__global__ void k_deg(const int* __restrict__ col, float* __restrict__ deg, int E) {
  int e = blockIdx.x * blockDim.x + threadIdx.x;
  if (e < E) atomicAdd(&deg[col[e]], 1.0f);
}

__global__ void k_rsqrt(float* d, int n) {
  int i = blockIdx.x * blockDim.x + threadIdx.x;
  if (i < n) d[i] = rsqrtf(d[i]);  // deg >= 1 always (self-loop)
}

// x (f32) -> bf16, 4 elements/thread, packed 8B stores.
__global__ void k_cast4(const float4* __restrict__ src, uint2* __restrict__ dst, long n4) {
  long q = (long)blockIdx.x * blockDim.x + threadIdx.x;
  if (q >= n4) return;
  float4 v = src[q];
  uint2 p;
  p.x = f2bfbits(v.x) | (f2bfbits(v.y) << 16);
  p.y = f2bfbits(v.z) | (f2bfbits(v.w) << 16);
  dst[q] = p;
}

// Swizzle W[128 x Cout] (f32, row-major) into per-lane B-fragment layout (bf16):
// fragment for (kk, half, col) occupies 16 contiguous bf16 (32B aligned):
//   jp<8 : K = kk*32 + half*8 + jp ; jp>=8 : K = kk*32 + 16 + half*8 + (jp-8)
__global__ void k_prep_w(const float* __restrict__ W, __bf16* __restrict__ Wsw, int Cout) {
  int i = blockIdx.x * blockDim.x + threadIdx.x;
  if (i >= 128 * Cout) return;
  int k = i / Cout, c = i % Cout;
  int kk = k >> 5, r = k & 31;
  int half = (r >> 3) & 1;
  int jp = (r & 16) ? (8 + (r & 7)) : (r & 7);
  Wsw[(((kk * 2 + half) * Cout) + c) * 16 + jp] = f2bf(W[i]);
}

// C[M x Cout] = A[M x 128](bf16, row-major) * W(pre-swizzled bf16) [+ bias]
// One block = one 16-row tile of A; one wave per 16-col tile of C.
// blockDim.x == 2*Cout (Cout/16 waves). M must be a multiple of 16.
__global__ void k_gemm_bf16(const __bf16* __restrict__ A, const __bf16* __restrict__ Wsw,
                            const float* __restrict__ bias, float* __restrict__ C,
                            int Cout) {
  __shared__ alignas(32) __bf16 sW[128 * 128];  // up to 32 KB, fragment layout
  __shared__ alignas(32) __bf16 sA[16 * 128];   // 4 KB, row-major tile
  const int tid = threadIdx.x;
  const int nthreads = blockDim.x;
  const int tile = blockIdx.x;

  // Async-stage W and the A tile into LDS (16B per lane per iteration).
  const unsigned wBytes = (unsigned)(128 * Cout * sizeof(__bf16));
  const unsigned ldsW = (unsigned)(size_t)&sW[0];
  const unsigned ldsA = (unsigned)(size_t)&sA[0];
  for (unsigned o = (unsigned)tid * 16u; o < wBytes; o += (unsigned)nthreads * 16u)
    async_copy_b128(Wsw, o, ldsW + o);
  const __bf16* Ab = A + (size_t)tile * 16 * 128;
  for (unsigned o = (unsigned)tid * 16u; o < 16u * 128u * (unsigned)sizeof(__bf16);
       o += (unsigned)nthreads * 16u)
    async_copy_b128(Ab, o, ldsA + o);
  wait_async0();
  __syncthreads();

  const int wave = tid >> 5;
  const int lane = tid & 31;
  const int m    = lane & 15;   // A row within tile
  const int half = lane >> 4;   // K-half selector
  const int n    = lane & 15;   // C col within wave tile
  const int colBase = wave * 16;

  // Hoisted, uniform-per-lane bias value (single predicated load, no per-store branch).
  const float bv = bias ? bias[colBase + n] : 0.0f;

  v8f c = {};
#pragma unroll
  for (int kk = 0; kk < 4; ++kk) {      // K = 4 x 32
    const int k0 = kk * 32 + half * 8;
    const v8bf alo = *(const v8bf*)&sA[m * 128 + k0];
    const v8bf ahi = *(const v8bf*)&sA[m * 128 + k0 + 16];
    const v16bf a = __builtin_shufflevector(alo, ahi, 0, 1, 2, 3, 4, 5, 6, 7,
                                            8, 9, 10, 11, 12, 13, 14, 15);
    const v16bf b = *(const v16bf*)&sW[(((kk * 2 + half) * Cout) + colBase + n) * 16];
    // (neg_a, A, neg_b, B, c_mod, C, reuse_a, reuse_b)
    c = __builtin_amdgcn_wmma_f32_16x16x32_bf16(false, a, false, b, (short)0, c,
                                                false, false);
  }

  float* Crow = C + (long)tile * 16 * Cout;
#pragma unroll
  for (int r = 0; r < 8; ++r) {
    const int mr = r + half * 8;
    Crow[mr * Cout + colBase + n] = c[r] + bv;
  }
}

// agg[col[e]] += t[row[e]] * dinv[row]*dinv[col] ; one wave per edge, float4/lane.
__global__ void k_scatter(const float* __restrict__ t, const int* __restrict__ row,
                          const int* __restrict__ col, const float* __restrict__ dinv,
                          float* __restrict__ agg, int E) {
  const int g = blockIdx.x * blockDim.x + threadIdx.x;
  const int e = g >> 5;
  if (e >= E) return;
  const int lane = g & 31;
  const int r = row[e];
  const int c = col[e];
  const float nrm = dinv[r] * dinv[c];
  const float4 v = ((const float4*)(t + (long)r * 128))[lane];
  float* dst = agg + (long)c * 128 + lane * 4;
  atomicAdd(dst + 0, v.x * nrm);
  atomicAdd(dst + 1, v.y * nrm);
  atomicAdd(dst + 2, v.z * nrm);
  atomicAdd(dst + 3, v.w * nrm);
}

// out = bf16(relu(agg + t*dinv^2 + bias)); 4 features/thread, packed 8B stores.
__global__ void k_finalize4(const float4* __restrict__ agg, const float4* __restrict__ t,
                            const float* __restrict__ dinv, const float4* __restrict__ bias4,
                            uint2* __restrict__ out, long n4) {
  long q = (long)blockIdx.x * blockDim.x + threadIdx.x;
  if (q >= n4) return;
  const float d = dinv[q >> 5];          // 32 float4-groups per node (128 feats)
  const float d2 = d * d;
  const float4 va = agg[q];
  const float4 vt = t[q];
  const float4 vb = bias4[q & 31];
  const float r0 = fmaxf(va.x + vt.x * d2 + vb.x, 0.f);
  const float r1 = fmaxf(va.y + vt.y * d2 + vb.y, 0.f);
  const float r2 = fmaxf(va.z + vt.z * d2 + vb.z, 0.f);
  const float r3 = fmaxf(va.w + vt.w * d2 + vb.w, 0.f);
  uint2 p;
  p.x = f2bfbits(r0) | (f2bfbits(r1) << 16);
  p.y = f2bfbits(r2) | (f2bfbits(r3) << 16);
  out[q] = p;
}

extern "C" void kernel_launch(void* const* d_in, const int* in_sizes, int n_in,
                              void* d_out, int out_size, void* d_ws, size_t ws_size,
                              hipStream_t stream) {
  const float* x  = (const float*)d_in[0];
  const int*   ei = (const int*)d_in[1];
  const float* W1 = (const float*)d_in[2];
  const float* b1 = (const float*)d_in[3];
  const float* W2 = (const float*)d_in[4];
  const float* b2 = (const float*)d_in[5];
  const float* Wl = (const float*)d_in[6];
  const float* bl = (const float*)d_in[7];
  float* out = (float*)d_out;

  const int N = in_sizes[0] / 128;   // 50000 (divisible by 16)
  const int E = in_sizes[1] / 2;     // 640000
  const int* row = ei;
  const int* col = ei + E;

  char* ws = (char*)d_ws;
  size_t off = 0;
  auto alloc = [&](size_t bytes) {
    char* p = ws + off;
    off = (off + bytes + 255) & ~(size_t)255;
    return p;
  };
  float*  dinv = (float*)alloc((size_t)N * sizeof(float));
  float*  t    = (float*)alloc((size_t)N * 128 * sizeof(float));
  float*  agg  = (float*)alloc((size_t)N * 128 * sizeof(float));
  __bf16* ab   = (__bf16*)alloc((size_t)N * 128 * sizeof(__bf16));
  __bf16* wsw1 = (__bf16*)alloc((size_t)128 * 128 * sizeof(__bf16));
  __bf16* wsw2 = (__bf16*)alloc((size_t)128 * 128 * sizeof(__bf16));
  __bf16* wswl = (__bf16*)alloc((size_t)128 * 64 * sizeof(__bf16));
  (void)ws_size; (void)n_in; (void)out_size;

  const long NF  = (long)N * 128;
  const long NF4 = NF / 4;
  auto cdiv = [](long a, long b) { return (unsigned)((a + b - 1) / b); };
  const dim3 blk(256);
  const unsigned gNF4 = cdiv(NF4, 256);
  const unsigned gE32 = cdiv((long)E * 32, 256);
  const unsigned gTiles = (unsigned)(N / 16);

  // degree -> dinv (self-loop => deg starts at 1)
  k_fill1 <<<cdiv(N, 256), blk, 0, stream>>>(dinv, N);
  k_deg   <<<cdiv(E, 256), blk, 0, stream>>>(col, dinv, E);
  k_rsqrt <<<cdiv(N, 256), blk, 0, stream>>>(dinv, N);

  // pre-swizzle weights into WMMA B-fragment layout (bf16)
  k_prep_w <<<cdiv(128 * 128, 256), blk, 0, stream>>>(W1, wsw1, 128);
  k_prep_w <<<cdiv(128 * 128, 256), blk, 0, stream>>>(W2, wsw2, 128);
  k_prep_w <<<cdiv(128 * 64, 256),  blk, 0, stream>>>(Wl, wswl, 64);

  // layer 1
  k_cast4     <<<gNF4, blk, 0, stream>>>((const float4*)x, (uint2*)ab, NF4);
  k_gemm_bf16 <<<gTiles, 256, 0, stream>>>(ab, wsw1, nullptr, t, 128);
  k_zero4     <<<gNF4, blk, 0, stream>>>((float4*)agg, NF4);
  k_scatter   <<<gE32, blk, 0, stream>>>(t, row, col, dinv, agg, E);
  k_finalize4 <<<gNF4, blk, 0, stream>>>((const float4*)agg, (const float4*)t, dinv,
                                         (const float4*)b1, (uint2*)ab, NF4);

  // layer 2
  k_gemm_bf16 <<<gTiles, 256, 0, stream>>>(ab, wsw2, nullptr, t, 128);
  k_zero4     <<<gNF4, blk, 0, stream>>>((float4*)agg, NF4);
  k_scatter   <<<gE32, blk, 0, stream>>>(t, row, col, dinv, agg, E);
  k_finalize4 <<<gNF4, blk, 0, stream>>>((const float4*)agg, (const float4*)t, dinv,
                                         (const float4*)b2, (uint2*)ab, NF4);

  // linear head: out = a2 @ Wl + bl  (64 cols -> 4 waves/block)
  k_gemm_bf16 <<<gTiles, 128, 0, stream>>>(ab, wswl, bl, out, 64);
}
